// KalmanSmartwatchModel_7679401525389
// MI455X (gfx1250) — compile-verified, matrix operand order checked by reading, not run on hardware
//
#include <hip/hip_runtime.h>
#include <hip/hip_bf16.h>

// ---------------------------------------------------------------------------
// EnKF smartwatch model for MI455X (gfx1250, wave32, WMMA).
// Heavy GEMMs (process/sensor MLPs) use v_wmma_f32_16x16x32_f16 with fp32
// accumulation. Weights pre-packed to CDNA5 B-matrix lane layout. Each wave
// owns a 32-row M-tile: inputs staged to LDS as f16 once, two accumulators
// share every B tile (halves L2 weight traffic), activations ping-pong in LDS.
// ---------------------------------------------------------------------------

typedef __attribute__((ext_vector_type(16))) _Float16 v16h;
typedef __attribute__((ext_vector_type(8)))  float    v8f;

#define NB   2048
#define NE   32
#define NROW (NB*NE)      // 65536
#define DX   14
#define DZ   14
#define KIN_PM 140        // WIN*DX
#define KIN_SM 220        // WIN*RAW

// ---------------------------------------------------------------------------
// Pack a row-major fp32 weight [K,N] into f16 WMMA B-tiles.
// Tile (kt,nt): lane l<16 -> col nt*16+l, K = kt*32 + {0..7, 16..23}
//              lane l>=16 -> col nt*16+(l-16), K = kt*32 + {8..15, 24..31}
// dst[((kt*ntiles+nt)*32 + lane)*16 + e]
// ---------------------------------------------------------------------------
__global__ void pack_b16(const float* __restrict__ W, _Float16* __restrict__ dst,
                         int K, int N, int ntiles)
{
    int tile = blockIdx.x;
    int lane = threadIdx.x & 31;
    int nt   = tile % ntiles;
    int kt   = tile / ntiles;
    int n    = nt * 16 + (lane & 15);
    int kb   = kt * 32 + ((lane & 16) ? 8 : 0);
    _Float16* out = dst + ((size_t)tile * 32 + lane) * 16;
#pragma unroll
    for (int e = 0; e < 8; ++e) {
        int k = kb + e;
        out[e] = (k < K && n < N) ? (_Float16)W[(size_t)k * N + n] : (_Float16)0.f;
    }
#pragma unroll
    for (int e = 0; e < 8; ++e) {
        int k = kb + 16 + e;
        out[8 + e] = (k < K && n < N) ? (_Float16)W[(size_t)k * N + n] : (_Float16)0.f;
    }
}

// A-tile loader from LDS row-major f16 activation (two contiguous 16B reads).
__device__ inline v16h load_a_lds(const _Float16* __restrict__ p)
{
    v16h a;
#pragma unroll
    for (int e = 0; e < 8; ++e) a[e] = p[e];
#pragma unroll
    for (int e = 0; e < 8; ++e) a[8 + e] = p[16 + e];
    return a;
}

__device__ inline float leaky(float x) { return fmaxf(x, 0.01f * x); }

#define WMMA_F16(a, b, c) \
    __builtin_amdgcn_wmma_f32_16x16x32_f16(false, (a), false, (b), (short)0, (c), false, false)

// ---------------------------------------------------------------------------
// Fused process-model MLP: 140 -> 256 (leaky) -> 512 (leaky) -> 14
// One wave (32 threads) per 32-row tile; 2048 blocks. Single-wave workgroup:
// barriers are hardware NOPs, LDS is wave-private.
// ---------------------------------------------------------------------------
__global__ __launch_bounds__(32) void pm_fused(
    const float* __restrict__ X,                 // [65536,140]
    const _Float16* __restrict__ W1,             // packed 5x16 tiles
    const _Float16* __restrict__ W3,             // packed 8x32 tiles
    const _Float16* __restrict__ Wm2,            // packed 16x1 tiles
    const float* __restrict__ b1, const float* __restrict__ b3,
    const float* __restrict__ bm2,
    float* __restrict__ pred)                    // [65536,14]
{
    __shared__ _Float16 act0[32][160];           // f16 input tile (K padded)
    __shared__ _Float16 act1[32][256];
    __shared__ _Float16 act2[32][512];
    const int lane  = threadIdx.x & 31;
    const int mrow  = lane & 15;
    const int khalf = (lane & 16) ? 8 : 0;       // K sub-base, also C-row base
    const int row0  = blockIdx.x * 32;

    // stage input tile to LDS (fp32 -> f16, zero-pad K to 160)
    for (int r = 0; r < 32; ++r) {
        const float* rp = X + (size_t)(row0 + r) * KIN_PM;
        for (int k = lane; k < 160; k += 32)
            act0[r][k] = (k < KIN_PM) ? (_Float16)rp[k] : (_Float16)0.f;
    }
    __syncthreads();

    // layer 1: 140(160) -> 256
    for (int nt = 0; nt < 16; ++nt) {
        v8f c0 = {}, c1 = {};
        for (int kt = 0; kt < 5; ++kt) {
            v16h a0 = load_a_lds(&act0[mrow][kt * 32 + khalf]);
            v16h a1 = load_a_lds(&act0[16 + mrow][kt * 32 + khalf]);
            v16h b  = *((const v16h*)W1 + (size_t)(kt * 16 + nt) * 32 + lane);
            c0 = WMMA_F16(a0, b, c0);
            c1 = WMMA_F16(a1, b, c1);
        }
        int n = nt * 16 + mrow;
        float bias = b1[n];
#pragma unroll
        for (int v = 0; v < 8; ++v) {
            act1[khalf + v][n]      = (_Float16)leaky(c0[v] + bias);
            act1[16 + khalf + v][n] = (_Float16)leaky(c1[v] + bias);
        }
    }
    __syncthreads();

    // layer 2: 256 -> 512
    for (int nt = 0; nt < 32; ++nt) {
        v8f c0 = {}, c1 = {};
        for (int kt = 0; kt < 8; ++kt) {
            v16h a0 = load_a_lds(&act1[mrow][kt * 32 + khalf]);
            v16h a1 = load_a_lds(&act1[16 + mrow][kt * 32 + khalf]);
            v16h b  = *((const v16h*)W3 + (size_t)(kt * 32 + nt) * 32 + lane);
            c0 = WMMA_F16(a0, b, c0);
            c1 = WMMA_F16(a1, b, c1);
        }
        int n = nt * 16 + mrow;
        float bias = b3[n];
#pragma unroll
        for (int v = 0; v < 8; ++v) {
            act2[khalf + v][n]      = (_Float16)leaky(c0[v] + bias);
            act2[16 + khalf + v][n] = (_Float16)leaky(c1[v] + bias);
        }
    }
    __syncthreads();

    // layer 3: 512 -> 14 (no activation)
    {
        v8f c0 = {}, c1 = {};
        for (int kt = 0; kt < 16; ++kt) {
            v16h a0 = load_a_lds(&act2[mrow][kt * 32 + khalf]);
            v16h a1 = load_a_lds(&act2[16 + mrow][kt * 32 + khalf]);
            v16h b  = *((const v16h*)Wm2 + (size_t)kt * 32 + lane);
            c0 = WMMA_F16(a0, b, c0);
            c1 = WMMA_F16(a1, b, c1);
        }
        int n = mrow;
        if (n < DX) {
            float bias = bm2[n];
#pragma unroll
            for (int v = 0; v < 8; ++v) {
                pred[(size_t)(row0 + khalf + v) * DX + n]      = c0[v] + bias;
                pred[(size_t)(row0 + 16 + khalf + v) * DX + n] = c1[v] + bias;
            }
        }
    }
}

// ---------------------------------------------------------------------------
// Fused sensor MLP on the 2048 distinct rows: 220->256->256->64->14
// One wave per 32-row tile; 64 blocks.
// ---------------------------------------------------------------------------
__global__ __launch_bounds__(32) void sm_fused(
    const float* __restrict__ X,                 // [2048,220]
    const _Float16* __restrict__ W2,             // 7x16 tiles
    const _Float16* __restrict__ W3,             // 8x16 tiles
    const _Float16* __restrict__ W5,             // 8x4 tiles
    const _Float16* __restrict__ W6,             // 2x1 tiles
    const float* __restrict__ b2, const float* __restrict__ b3,
    const float* __restrict__ b5, const float* __restrict__ b6,
    float* __restrict__ sens)                    // [2048,14]
{
    __shared__ _Float16 act0[32][224];
    __shared__ _Float16 act1[32][256];
    __shared__ _Float16 act2[32][256];
    __shared__ _Float16 act3[32][64];
    const int lane  = threadIdx.x & 31;
    const int mrow  = lane & 15;
    const int khalf = (lane & 16) ? 8 : 0;
    const int row0  = blockIdx.x * 32;

    for (int r = 0; r < 32; ++r) {
        const float* rp = X + (size_t)(row0 + r) * KIN_SM;
        for (int k = lane; k < 224; k += 32)
            act0[r][k] = (k < KIN_SM) ? (_Float16)rp[k] : (_Float16)0.f;
    }
    __syncthreads();

    for (int nt = 0; nt < 16; ++nt) {            // 220(224) -> 256
        v8f c0 = {}, c1 = {};
        for (int kt = 0; kt < 7; ++kt) {
            v16h a0 = load_a_lds(&act0[mrow][kt * 32 + khalf]);
            v16h a1 = load_a_lds(&act0[16 + mrow][kt * 32 + khalf]);
            v16h b  = *((const v16h*)W2 + (size_t)(kt * 16 + nt) * 32 + lane);
            c0 = WMMA_F16(a0, b, c0);
            c1 = WMMA_F16(a1, b, c1);
        }
        int n = nt * 16 + mrow;
        float bias = b2[n];
#pragma unroll
        for (int v = 0; v < 8; ++v) {
            act1[khalf + v][n]      = (_Float16)leaky(c0[v] + bias);
            act1[16 + khalf + v][n] = (_Float16)leaky(c1[v] + bias);
        }
    }
    __syncthreads();

    for (int nt = 0; nt < 16; ++nt) {            // 256 -> 256
        v8f c0 = {}, c1 = {};
        for (int kt = 0; kt < 8; ++kt) {
            v16h a0 = load_a_lds(&act1[mrow][kt * 32 + khalf]);
            v16h a1 = load_a_lds(&act1[16 + mrow][kt * 32 + khalf]);
            v16h b  = *((const v16h*)W3 + (size_t)(kt * 16 + nt) * 32 + lane);
            c0 = WMMA_F16(a0, b, c0);
            c1 = WMMA_F16(a1, b, c1);
        }
        int n = nt * 16 + mrow;
        float bias = b3[n];
#pragma unroll
        for (int v = 0; v < 8; ++v) {
            act2[khalf + v][n]      = (_Float16)leaky(c0[v] + bias);
            act2[16 + khalf + v][n] = (_Float16)leaky(c1[v] + bias);
        }
    }
    __syncthreads();

    for (int nt = 0; nt < 4; ++nt) {             // 256 -> 64
        v8f c0 = {}, c1 = {};
        for (int kt = 0; kt < 8; ++kt) {
            v16h a0 = load_a_lds(&act2[mrow][kt * 32 + khalf]);
            v16h a1 = load_a_lds(&act2[16 + mrow][kt * 32 + khalf]);
            v16h b  = *((const v16h*)W5 + (size_t)(kt * 4 + nt) * 32 + lane);
            c0 = WMMA_F16(a0, b, c0);
            c1 = WMMA_F16(a1, b, c1);
        }
        int n = nt * 16 + mrow;
        float bias = b5[n];
#pragma unroll
        for (int v = 0; v < 8; ++v) {
            act3[khalf + v][n]      = (_Float16)leaky(c0[v] + bias);
            act3[16 + khalf + v][n] = (_Float16)leaky(c1[v] + bias);
        }
    }
    __syncthreads();

    {                                            // 64 -> 14 (no activation)
        v8f c0 = {}, c1 = {};
        for (int kt = 0; kt < 2; ++kt) {
            v16h a0 = load_a_lds(&act3[mrow][kt * 32 + khalf]);
            v16h a1 = load_a_lds(&act3[16 + mrow][kt * 32 + khalf]);
            v16h b  = *((const v16h*)W6 + (size_t)kt * 32 + lane);
            c0 = WMMA_F16(a0, b, c0);
            c1 = WMMA_F16(a1, b, c1);
        }
        int n = mrow;
        if (n < DZ) {
            float bias = b6[n];
#pragma unroll
            for (int v = 0; v < 8; ++v) {
                sens[(size_t)(row0 + khalf + v) * DZ + n]      = c0[v] + bias;
                sens[(size_t)(row0 + 16 + khalf + v) * DZ + n] = c1[v] + bias;
            }
        }
    }
}

// ensemble_z[b,e,:] = sensor_out[(b*32+e) & 2047, :]
__global__ void ensz_kernel(const float* __restrict__ sens, float* __restrict__ out_ensz)
{
    int idx = blockIdx.x * blockDim.x + threadIdx.x;
    if (idx >= NROW * DZ) return;
    int r = idx / DZ, d = idx - r * DZ;
    out_ensz[idx] = sens[(size_t)(r & (NB - 1)) * DZ + d];
}

// state_m[b,d] and z[b,d] = ensemble means
__global__ void means_kernel(const float* __restrict__ pred, const float* __restrict__ sens,
                             float* __restrict__ out_state_m, float* __restrict__ out_z)
{
    int idx = blockIdx.x * blockDim.x + threadIdx.x;
    if (idx >= NB * DX) return;
    int b = idx / DX, d = idx - b * DX;
    float sp = 0.f, sz = 0.f;
    for (int e = 0; e < NE; ++e) {
        int r = b * NE + e;
        sp += pred[(size_t)r * DX + d];
        sz += sens[(size_t)(r & (NB - 1)) * DZ + d];
    }
    out_state_m[idx] = sp * (1.f / NE);
    out_z[idx]       = sz * (1.f / NE);
}

// R diagonal: r = (relu(z@W1+b1)@W2 + b2 + 0.001)^2 + 0.038729833
__global__ void obs_noise(const float* __restrict__ z,
                          const float* __restrict__ w1, const float* __restrict__ bb1,
                          const float* __restrict__ w2, const float* __restrict__ bb2,
                          float* __restrict__ rdiag)
{
    int b = blockIdx.x * blockDim.x + threadIdx.x;
    if (b >= NB) return;
    float zz[DZ];
#pragma unroll
    for (int d = 0; d < DZ; ++d) zz[d] = z[(size_t)b * DZ + d];
    float h[32];
#pragma unroll 4
    for (int j = 0; j < 32; ++j) {
        float s = bb1[j];
        for (int d = 0; d < DZ; ++d) s += zz[d] * w1[d * 32 + j];
        h[j] = (s > 0.f) ? s : 0.f;
    }
#pragma unroll 2
    for (int d = 0; d < DZ; ++d) {
        float s = bb2[d];
        for (int j = 0; j < 32; ++j) s += h[j] * w2[j * DZ + d];
        s += 0.001f;
        rdiag[(size_t)b * DZ + d] = s * s + 0.038729833f;
    }
}

// Per-batch EnKF correction: S = A^T A, inv(S/31 + R), K, gain, outputs.
__global__ __launch_bounds__(64) void enkf_kernel(
    const float* __restrict__ pred,      // [B*E,14]
    const float* __restrict__ state_m,   // [B,14]
    const float* __restrict__ ens_z,     // [B*E,14]
    const float* __restrict__ rdiag,     // [B,14]
    float* __restrict__ out_corr,        // [B*E,14]
    float* __restrict__ out_corr_mean)   // [B,14]
{
    __shared__ float P[NE][DX];
    __shared__ float A[NE][DX];
    __shared__ float M1[DX][DX];
    __shared__ float G[DX][2 * DX];
    __shared__ float Km[DX][DX];
    __shared__ float fcol[DX];
    __shared__ float corr[NE][DX];
    const int b = blockIdx.x;
    const int t = threadIdx.x;

    for (int i = t; i < NE * DX; i += 64) {
        int e = i / DX, d = i - e * DX;
        float p = pred[((size_t)b * NE + e) * DX + d];
        P[e][d] = p;
        A[e][d] = p - state_m[(size_t)b * DX + d];
    }
    __syncthreads();
    for (int i = t; i < DX * DX; i += 64) {
        int r = i / DX, c = i - r * DX;
        float s = 0.f;
        for (int e = 0; e < NE; ++e) s += A[e][r] * A[e][c];
        M1[r][c] = s * (1.f / (NE - 1));
    }
    __syncthreads();
    for (int i = t; i < DX * 2 * DX; i += 64) {
        int r = i / (2 * DX), c = i - r * 2 * DX;
        float v;
        if (c < DX) v = M1[r][c] + ((c == r) ? rdiag[(size_t)b * DX + r] : 0.f);
        else        v = (c - DX == r) ? 1.f : 0.f;
        G[r][c] = v;
    }
    __syncthreads();
    // Gauss-Jordan (innovation is SPD: no pivoting needed)
    for (int k = 0; k < DX; ++k) {
        float piv = G[k][k];
        __syncthreads();
        float pinv = 1.f / piv;
        for (int c = t; c < 2 * DX; c += 64) G[k][c] *= pinv;
        if (t < DX) fcol[t] = (t == k) ? 0.f : G[t][k];
        __syncthreads();
        for (int i = t; i < DX * 2 * DX; i += 64) {
            int r = i / (2 * DX), c = i - r * 2 * DX;
            if (r != k) G[r][c] -= fcol[r] * G[k][c];
        }
        __syncthreads();
    }
    // K = M1 @ inv(innovation)
    for (int i = t; i < DX * DX; i += 64) {
        int r = i / DX, c = i - r * DX;
        float s = 0.f;
        for (int l = 0; l < DX; ++l) s += M1[r][l] * G[l][DX + c];
        Km[r][c] = s;
    }
    __syncthreads();
    // corrected = pred + K @ (ens_z - pred)
    for (int i = t; i < NE * DX; i += 64) {
        int e = i / DX, r = i - e * DX;
        float g = 0.f;
        for (int j = 0; j < DZ; ++j)
            g += Km[r][j] * (ens_z[((size_t)b * NE + e) * DZ + j] - P[e][j]);
        float val = P[e][r] + g;
        corr[e][r] = val;
        out_corr[((size_t)b * NE + e) * DX + r] = val;
    }
    __syncthreads();
    if (t < DX) {
        float s = 0.f;
        for (int e = 0; e < NE; ++e) s += corr[e][t];
        out_corr_mean[(size_t)b * DX + t] = s * (1.f / NE);
    }
}

// ---------------------------------------------------------------------------
extern "C" void kernel_launch(void* const* d_in, const int* in_sizes, int n_in,
                              void* d_out, int out_size, void* d_ws, size_t ws_size,
                              hipStream_t stream)
{
    const float* raw_obs = (const float*)d_in[0];
    const float* st_prev = (const float*)d_in[1];
    const float* pm_w1   = (const float*)d_in[2];
    const float* pm_b1   = (const float*)d_in[3];
    const float* pm_w3   = (const float*)d_in[4];
    const float* pm_b3   = (const float*)d_in[5];
    const float* pm_wm2  = (const float*)d_in[6];
    const float* pm_bm2  = (const float*)d_in[7];
    const float* sm_w2   = (const float*)d_in[8];
    const float* sm_b2   = (const float*)d_in[9];
    const float* sm_w3   = (const float*)d_in[10];
    const float* sm_b3   = (const float*)d_in[11];
    const float* sm_w5   = (const float*)d_in[12];
    const float* sm_b5   = (const float*)d_in[13];
    const float* sm_w6   = (const float*)d_in[14];
    const float* sm_b6   = (const float*)d_in[15];
    const float* on_w1   = (const float*)d_in[16];
    const float* on_b1   = (const float*)d_in[17];
    const float* on_w2   = (const float*)d_in[18];
    const float* on_b2   = (const float*)d_in[19];

    float* out  = (float*)d_out;
    float* out0 = out;                       // state_corrected [B,E,14]
    float* out1 = out0 + (size_t)NROW * DX;  // corrected mean  [B,1,14]
    float* out2 = out1 + (size_t)NB * DX;    // state_m         [B,1,14]
    float* out3 = out2 + (size_t)NB * DX;    // z               [B,1,14]
    float* out4 = out3 + (size_t)NB * DZ;    // ensemble_z      [B,E,14]

    char* ws = (char*)d_ws;
    size_t off = 0;
    auto alloc = [&](size_t bytes) -> void* {
        void* p = ws + off;
        off = (off + bytes + 255) & ~(size_t)255;
        return p;
    };
    _Float16* W1h  = (_Float16*)alloc((size_t)5  * 16 * 512 * 2);
    _Float16* W3h  = (_Float16*)alloc((size_t)8  * 32 * 512 * 2);
    _Float16* Wm2h = (_Float16*)alloc((size_t)16 * 1  * 512 * 2);
    _Float16* SW2h = (_Float16*)alloc((size_t)7  * 16 * 512 * 2);
    _Float16* SW3h = (_Float16*)alloc((size_t)8  * 16 * 512 * 2);
    _Float16* SW5h = (_Float16*)alloc((size_t)8  * 4  * 512 * 2);
    _Float16* SW6h = (_Float16*)alloc((size_t)2  * 1  * 512 * 2);
    float* pred   = (float*)alloc((size_t)NROW * DX * 4);
    float* sens   = (float*)alloc((size_t)NB * DZ * 4);
    float* rdiag  = (float*)alloc((size_t)NB * DZ * 4);

    // weight packing (trivial cost)
    pack_b16<<<5 * 16, 32, 0, stream>>>(pm_w1,  W1h,  140, 256, 16);
    pack_b16<<<8 * 32, 32, 0, stream>>>(pm_w3,  W3h,  256, 512, 32);
    pack_b16<<<16,     32, 0, stream>>>(pm_wm2, Wm2h, 512, 14,  1);
    pack_b16<<<7 * 16, 32, 0, stream>>>(sm_w2,  SW2h, 220, 256, 16);
    pack_b16<<<8 * 16, 32, 0, stream>>>(sm_w3,  SW3h, 256, 256, 16);
    pack_b16<<<8 * 4,  32, 0, stream>>>(sm_w5,  SW5h, 256, 64,  4);
    pack_b16<<<2,      32, 0, stream>>>(sm_w6,  SW6h, 64,  14,  1);

    // heavy WMMA MLPs (one wave per 32-row tile)
    pm_fused<<<NROW / 32, 32, 0, stream>>>(st_prev, W1h, W3h, Wm2h,
                                           pm_b1, pm_b3, pm_bm2, pred);
    sm_fused<<<NB / 32, 32, 0, stream>>>(raw_obs, SW2h, SW3h, SW5h, SW6h,
                                         sm_b2, sm_b3, sm_b5, sm_b6, sens);

    // small fp32 tail
    ensz_kernel <<<(NROW * DZ) / 256, 256, 0, stream>>>(sens, out4);
    means_kernel<<<(NB * DX) / 256,   256, 0, stream>>>(pred, sens, out2, out3);
    obs_noise   <<<NB / 256,          256, 0, stream>>>(out3, on_w1, on_b1, on_w2, on_b2, rdiag);
    enkf_kernel <<<NB, 64, 0, stream>>>(pred, out2, out4, rdiag, out0, out1);
}
